// Transformer_78469052498467
// MI455X (gfx1250) — matrix operand order repaired and checked by reading, not compiled, not run on hardware
//
#include <hip/hip_runtime.h>

typedef __attribute__((ext_vector_type(16))) _Float16 v16h;
typedef __attribute__((ext_vector_type(8)))  _Float16 v8h;
typedef __attribute__((ext_vector_type(8)))  float    v8f;

#define BM 128
#define BN 64
#define BK 32

// ---------------- WMMA fragment loaders (contiguous k = 16*(lane/16) + j) ----------------
// A fragment: 16x32 f16, tile row-major with row stride BK.
__device__ inline v16h frag_a(const _Float16* t, int lane) {
  int m  = lane & 15;
  int kb = (lane < 16) ? 0 : 16;
  v16h f;
#pragma unroll
  for (int j = 0; j < 16; ++j) {
    int k = kb + j;
    f[j] = t[m * BK + k];
  }
  return f;
}
// B fragment: 32x16 (KxN); W tile stored as [N][K] row-major (W is N x K).
__device__ inline v16h frag_b(const _Float16* t, int lane) {
  int n  = lane & 15;
  int kb = (lane < 16) ? 0 : 16;
  v16h f;
#pragma unroll
  for (int j = 0; j < 16; ++j) f[j] = t[n * BK + kb + j];
  return f;
}

// Branch-free fetch of 8 consecutive floats (row-clamped) converted to 8 halves.
__device__ inline v8h fetch8_h(const float* __restrict__ base, long rowStride,
                               int row, int rowMax, int colOff) {
  bool ok = row < rowMax;
  const float* p = base + (long)(ok ? row : 0) * rowStride + colOff;
  float4 a0 = *(const float4*)(p);
  float4 a1 = *(const float4*)(p + 4);
  v8h h;
  h[0] = (_Float16)(ok ? a0.x : 0.0f);
  h[1] = (_Float16)(ok ? a0.y : 0.0f);
  h[2] = (_Float16)(ok ? a0.z : 0.0f);
  h[3] = (_Float16)(ok ? a0.w : 0.0f);
  h[4] = (_Float16)(ok ? a1.x : 0.0f);
  h[5] = (_Float16)(ok ? a1.y : 0.0f);
  h[6] = (_Float16)(ok ? a1.z : 0.0f);
  h[7] = (_Float16)(ok ? a1.w : 0.0f);
  return h;
}

// ---------------- Tiled WMMA GEMM: C = act(A @ W^T + bias) ------------------
// A: MxK fp32 row-major, W: NxK fp32 row-major. act: 0=none 1=relu 2=sigmoid.
// Output fp32 to Cf, or f16 to Ch (if Ch != nullptr). 256 thr = 8 waves;
// wave w computes rows [w*16, w*16+16) x 64 cols (4 WMMA tiles).
__global__ __launch_bounds__(256)
void gemm_wmma_kernel(const float* __restrict__ A, const float* __restrict__ W,
                      const float* __restrict__ bias, float* __restrict__ Cf,
                      _Float16* __restrict__ Ch, int M, int N, int K, int act) {
  __shared__ _Float16 As[BM * BK];   // 8 KB
  __shared__ _Float16 Bs[BN * BK];   // 4 KB
  int bn = blockIdx.x * BN;
  int bm = blockIdx.y * BM;
  int tid  = threadIdx.x;
  int lane = tid & 31;
  int wave = tid >> 5;   // 0..7 -> 16-row slice

  // load mapping: chunks of 8 floats along K
  int ar0 = tid >> 2;            // A chunk 0 row (0..63)
  int ac  = (tid & 3) * 8;       // k sub-offset 0/8/16/24
  int br  = tid >> 2;            // B row (0..63)

  v8f c0 = {}, c1 = {}, c2 = {}, c3 = {};

  for (int k0 = 0; k0 < K; k0 += BK) {
    v8h ha0 = fetch8_h(A, K, bm + ar0,      M, k0 + ac);
    v8h ha1 = fetch8_h(A, K, bm + ar0 + 64, M, k0 + ac);
    v8h hb  = fetch8_h(W, K, bn + br,       N, k0 + ac);
    *(v8h*)(&As[ar0 * BK + ac])        = ha0;
    *(v8h*)(&As[(ar0 + 64) * BK + ac]) = ha1;
    *(v8h*)(&Bs[br * BK + ac])         = hb;
    if (k0 + BK < K) {  // prefetch next K tile (global_prefetch_b8)
      __builtin_prefetch(A + (long)(bm + ar0) * K + k0 + BK + ac, 0, 1);
      __builtin_prefetch(W + (long)(bn + br) * K + k0 + BK + ac, 0, 1);
    }
    __syncthreads();
    v16h af = frag_a(&As[(wave * 16) * BK], lane);
    v16h b0 = frag_b(&Bs[0 * BK],  lane);
    v16h b1 = frag_b(&Bs[16 * BK], lane);
    v16h b2 = frag_b(&Bs[32 * BK], lane);
    v16h b3 = frag_b(&Bs[48 * BK], lane);
    c0 = __builtin_amdgcn_wmma_f32_16x16x32_f16(false, af, false, b0, (short)0, c0, false, false);
    c1 = __builtin_amdgcn_wmma_f32_16x16x32_f16(false, af, false, b1, (short)0, c1, false, false);
    c2 = __builtin_amdgcn_wmma_f32_16x16x32_f16(false, af, false, b2, (short)0, c2, false, false);
    c3 = __builtin_amdgcn_wmma_f32_16x16x32_f16(false, af, false, b3, (short)0, c3, false, false);
    __syncthreads();
  }

  int rowBase = bm + wave * 16 + ((lane < 16) ? 0 : 8);
  int colLane = bn + (lane & 15);
#pragma unroll
  for (int t = 0; t < 4; ++t) {
    int col = colLane + t * 16;
    if (col >= N) continue;
    float bv = bias ? bias[col] : 0.0f;
    v8f cc = (t == 0) ? c0 : (t == 1) ? c1 : (t == 2) ? c2 : c3;
#pragma unroll
    for (int v = 0; v < 8; ++v) {
      int row = rowBase + v;
      if (row >= M) continue;
      float x = cc[v] + bv;
      if (act == 1) x = fmaxf(x, 0.0f);
      else if (act == 2) x = 1.0f / (1.0f + __expf(-x));
      if (Ch) Ch[(long)row * N + col] = (_Float16)x;
      else    Cf[(long)row * N + col] = x;
    }
  }
}

// ---------------- elementwise add ----------------
__global__ void add_kernel(const float* __restrict__ a, const float* __restrict__ b,
                           float* __restrict__ o, long n) {
  long i = (long)blockIdx.x * 256 + threadIdx.x;
  if (i < n) o[i] = a[i] + b[i];
}

// ---------------- small multi-head attention (dh=32, H=8) ----------------
// qkv: (nb, seq, 768) [q|k|v each 256 = 8 heads * 32]; out: (nb, seq, 256)
__global__ __launch_bounds__(128)
void attn_kernel(const float* __restrict__ qkv, float* __restrict__ out,
                 int seq, int nb) {
  __shared__ float Ks[100 * 32];
  __shared__ float Vs[100 * 32];
  int b = blockIdx.x >> 3;
  int h = blockIdx.x & 7;
  if (b >= nb) return;
  int tid = threadIdx.x;
  for (int idx = tid; idx < seq * 32; idx += 128) {
    int s = idx >> 5, c = idx & 31;
    long base = ((long)b * seq + s) * 768 + h * 32 + c;
    Ks[idx] = qkv[base + 256];
    Vs[idx] = qkv[base + 512];
  }
  __syncthreads();
  int qi = tid;
  if (qi >= seq) return;
  const float scale = 0.17677669529663687f;  // 1/sqrt(32)
  float q[32];
  const float* qrow = qkv + ((long)b * seq + qi) * 768 + h * 32;
#pragma unroll
  for (int c = 0; c < 32; ++c) q[c] = qrow[c];
  float mx = -1e30f;
  for (int j = 0; j < seq; ++j) {
    float s = 0.0f;
#pragma unroll
    for (int c = 0; c < 32; ++c) s += q[c] * Ks[j * 32 + c];
    s *= scale;
    mx = fmaxf(mx, s);
  }
  float l = 0.0f;
  float o[32];
#pragma unroll
  for (int c = 0; c < 32; ++c) o[c] = 0.0f;
  for (int j = 0; j < seq; ++j) {
    float s = 0.0f;
#pragma unroll
    for (int c = 0; c < 32; ++c) s += q[c] * Ks[j * 32 + c];
    s *= scale;
    float p = __expf(s - mx);
    l += p;
#pragma unroll
    for (int c = 0; c < 32; ++c) o[c] += p * Vs[j * 32 + c];
  }
  float inv = 1.0f / l;
  float* orow = out + ((long)b * seq + qi) * 256 + h * 32;
#pragma unroll
  for (int c = 0; c < 32; ++c) orow[c] = o[c] * inv;
}

// ---------------- residual / gate / clip + LayerNorm (block per row of 256) --
// mode 0: LN(a+b)   mode 1: LN(g1*a+g2*b), g row of 512 from grow
// mode 2: LN(clip(a+b, +-65504))
__global__ __launch_bounds__(256)
void ln_kernel(const float* __restrict__ a, const float* __restrict__ b,
               const float* __restrict__ grow, const float* __restrict__ gamma,
               const float* __restrict__ beta, float* __restrict__ out, int mode) {
  __shared__ float red[256];
  int row = blockIdx.x;
  int c = threadIdx.x;
  long base = (long)row * 256;
  float x;
  if (mode == 1) {
    float g1 = grow[(long)row * 512 + c];
    float g2 = grow[(long)row * 512 + 256 + c];
    x = g1 * a[base + c] + g2 * b[base + c];
  } else {
    x = a[base + c] + b[base + c];
    if (mode == 2) x = fminf(fmaxf(x, -65504.0f), 65504.0f);
  }
  red[c] = x;
  __syncthreads();
  for (int s = 128; s > 0; s >>= 1) { if (c < s) red[c] += red[c + s]; __syncthreads(); }
  float mean = red[0] * (1.0f / 256.0f);
  __syncthreads();
  float d = x - mean;
  red[c] = d * d;
  __syncthreads();
  for (int s = 128; s > 0; s >>= 1) { if (c < s) red[c] += red[c + s]; __syncthreads(); }
  float var = red[0] * (1.0f / 256.0f);
  out[base + c] = d * rsqrtf(var + 1e-5f) * gamma[c] + beta[c];
}

// ---------------- (b,nq,K,d) <-> (b,K,nq,d) transpose ----------------
__global__ void transpose_kernel(const float* __restrict__ in, float* __restrict__ out, int dir) {
  long i = (long)blockIdx.x * 256 + threadIdx.x;
  const long N = (long)8 * 100 * 17 * 256;
  if (i >= N) return;
  int c = (int)(i & 255);
  long r = i >> 8;
  int kk = (int)(r % 17);
  long r2 = r / 17;
  int q = (int)(r2 % 100);
  int b = (int)(r2 / 100);
  long dst = (((long)b * 17 + kk) * 100 + q) * 256 + c;
  if (dir == 0) out[dst] = in[i];
  else          out[i]   = in[dst];
}

// ---------------- softmax over 16 (L*P) per (row, head) ----------------
__global__ void softmax16_kernel(float* __restrict__ aw, int units) {
  int u = blockIdx.x * 256 + threadIdx.x;
  if (u >= units) return;
  int r = u >> 3, h = u & 7;
  float* p = aw + (long)r * 128 + h * 16;
  float m = -1e30f;
#pragma unroll
  for (int i = 0; i < 16; ++i) m = fmaxf(m, p[i]);
  float e[16], s = 0.0f;
#pragma unroll
  for (int i = 0; i < 16; ++i) { e[i] = __expf(p[i] - m); s += e[i]; }
  float inv = 1.0f / s;
#pragma unroll
  for (int i = 0; i < 16; ++i) p[i] = e[i] * inv;
}

// ---------------- concat [a|b] into 512-wide rows ----------------
__global__ void concat_kernel(const float* __restrict__ a, const float* __restrict__ b,
                              float* __restrict__ o, long n) {
  long i = (long)blockIdx.x * 256 + threadIdx.x;
  if (i >= n) return;
  int c = (int)(i & 511);
  long r = i >> 9;
  o[i] = (c < 256) ? a[r * 256 + c] : b[r * 256 + (c - 256)];
}

// ---------------- MS-deformable bilinear sampling ----------------
// One wave per (b, lq, head); lane = channel (dh=32). value is f16 (B,34000,8,32).
__global__ __launch_bounds__(256)
void deform_kernel(const float* __restrict__ ref, const float* __restrict__ offs,
                   const float* __restrict__ aw, const _Float16* __restrict__ value,
                   float* __restrict__ out) {
  int gid = blockIdx.x * 8 + (threadIdx.x >> 5);
  int lane = threadIdx.x & 31;
  if (gid >= 8 * 1700 * 8) return;
  int h  = gid & 7;
  int bl = gid >> 3;            // b*1700 + lq
  int b  = bl / 1700;
  const int lvl_h[4]     = {160, 80, 40, 20};
  const int lvl_w[4]     = {160, 80, 40, 20};
  const int lvl_start[4] = {0, 25600, 32000, 33600};
  const float* offrow = offs + (long)bl * 256;
  const float* awrow  = aw   + (long)bl * 128 + h * 16;
  const float* refrow = ref  + (long)bl * 8;
  long vb = (long)b * 34000;
  float o = 0.0f;
#pragma unroll
  for (int l = 0; l < 4; ++l) {
    float rx = refrow[l * 2 + 0], ry = refrow[l * 2 + 1];
    int hh = lvl_h[l], ww = lvl_w[l];
    float fw = (float)ww, fh = (float)hh;
    long vstart = (vb + lvl_start[l]) * 256 + h * 32 + lane;
#pragma unroll
    for (int p = 0; p < 4; ++p) {
      int oi = (h * 16 + l * 4 + p) * 2;
      float px = (rx + offrow[oi]     / fw) * fw - 0.5f;
      float py = (ry + offrow[oi + 1] / fh) * fh - 0.5f;
      float x0f = floorf(px), y0f = floorf(py);
      float lx = px - x0f, ly = py - y0f;
      int x0 = (int)x0f, y0 = (int)y0f;
      float a = awrow[l * 4 + p];
      float s = 0.0f;
#pragma unroll
      for (int dy = 0; dy < 2; ++dy) {
#pragma unroll
        for (int dx = 0; dx < 2; ++dx) {
          int xi = x0 + dx, yi = y0 + dy;
          float wgt = (dx ? lx : 1.0f - lx) * (dy ? ly : 1.0f - ly);
          bool valid = (xi >= 0) && (xi < ww) && (yi >= 0) && (yi < hh);
          int xc = xi < 0 ? 0 : (xi > ww - 1 ? ww - 1 : xi);
          int yc = yi < 0 ? 0 : (yi > hh - 1 ? hh - 1 : yi);
          float g = (float)value[vstart + (long)(yc * ww + xc) * 256];
          s += valid ? wgt * g : 0.0f;
        }
      }
      o += a * s;
    }
  }
  out[(long)bl * 256 + h * 32 + lane] = o;
}

// ======================= host launcher =======================
extern "C" void kernel_launch(void* const* d_in, const int* in_sizes, int n_in,
                              void* d_out, int out_size, void* d_ws, size_t ws_size,
                              hipStream_t stream) {
  (void)in_sizes; (void)n_in; (void)out_size; (void)ws_size;
  const float* tgt_pose = (const float*)d_in[0];
  const float* pos      = (const float*)d_in[1];
  const float* refpts   = (const float*)d_in[2];
  const float* memory   = (const float*)d_in[3];
  const float* win_Wqkv = (const float*)d_in[4];
  const float* win_bqkv = (const float*)d_in[5];
  const float* win_Wo   = (const float*)d_in[6];
  const float* win_bo   = (const float*)d_in[7];
  const float* win_ng   = (const float*)d_in[8];
  const float* win_nb   = (const float*)d_in[9];
  const float* acr_Wqkv = (const float*)d_in[10];
  const float* acr_bqkv = (const float*)d_in[11];
  const float* acr_Wo   = (const float*)d_in[12];
  const float* acr_bo   = (const float*)d_in[13];
  const float* acr_ng   = (const float*)d_in[14];
  const float* acr_nb   = (const float*)d_in[15];
  const float* off_W    = (const float*)d_in[16];
  const float* off_b    = (const float*)d_in[17];
  const float* aw_W     = (const float*)d_in[18];
  const float* aw_b     = (const float*)d_in[19];
  const float* val_W    = (const float*)d_in[20];
  const float* val_b    = (const float*)d_in[21];
  const float* outp_W   = (const float*)d_in[22];
  const float* outp_b   = (const float*)d_in[23];
  const float* gate_W   = (const float*)d_in[24];
  const float* gate_b   = (const float*)d_in[25];
  const float* gate_ng  = (const float*)d_in[26];
  const float* gate_nb  = (const float*)d_in[27];
  const float* ffn_W1   = (const float*)d_in[28];
  const float* ffn_b1   = (const float*)d_in[29];
  const float* ffn_W2   = (const float*)d_in[30];
  const float* ffn_b2   = (const float*)d_in[31];
  const float* n2_g     = (const float*)d_in[32];
  const float* n2_b     = (const float*)d_in[33];

  // Workspace layout (float units). VAL span (34,816,000 floats holding f16)
  // is reused for CAT/G/H scratch after the sampler consumes it.
  float* ws   = (float*)d_ws;
  float* TGT  = ws + 0;           // 3,481,600
  float* TGTQ = ws + 3481600;     // 3,481,600
  float* X    = ws + 6963200;     // 3,481,600
  float* T2   = ws + 10444800;    // 3,481,600
  float* ATT  = ws + 13926400;    // 3,481,600
  float* QKV  = ws + 17408000;    // 10,444,800
  float* AW   = ws + 27852800;    // 1,740,800
  float* OFFS = ws + 29593600;    // 3,481,600
  _Float16* VALH = (_Float16*)(ws + 33075200);  // 69,632,000 halves
  float* CAT  = ws + 33075200;    // reuse of VAL span
  float* G    = ws + 40038400;
  float* H    = ws + 47001600;    // 13,926,400 (ends at 60,928,000 floats)

  const int M1 = 13600;
  const long nE = (long)M1 * 256;

  auto gemm = [&](const float* A, const float* W, const float* bias,
                  float* Cf, _Float16* Ch, int M, int N, int K, int act) {
    dim3 g((N + BN - 1) / BN, (M + BM - 1) / BM);
    gemm_wmma_kernel<<<g, 256, 0, stream>>>(A, W, bias, Cf, Ch, M, N, K, act);
  };

  // tgt = tgt_pose + pos
  add_kernel<<<(int)((nE + 255) / 256), 256, 0, stream>>>(tgt_pose, pos, TGT, nE);
  // within-keypoint MHA
  gemm(TGT, win_Wqkv, win_bqkv, QKV, nullptr, M1, 768, 256, 0);
  attn_kernel<<<800 * 8, 128, 0, stream>>>(QKV, ATT, 17, 800);
  gemm(ATT, win_Wo, win_bo, T2, nullptr, M1, 256, 256, 0);
  ln_kernel<<<M1, 256, 0, stream>>>(TGT, T2, nullptr, win_ng, win_nb, TGT, 0);
  // across-query MHA
  transpose_kernel<<<(int)((nE + 255) / 256), 256, 0, stream>>>(TGT, X, 0);
  gemm(X, acr_Wqkv, acr_bqkv, QKV, nullptr, M1, 768, 256, 0);
  attn_kernel<<<136 * 8, 128, 0, stream>>>(QKV, ATT, 100, 136);
  gemm(ATT, acr_Wo, acr_bo, T2, nullptr, M1, 256, 256, 0);
  ln_kernel<<<M1, 256, 0, stream>>>(X, T2, nullptr, acr_ng, acr_nb, X, 0);
  transpose_kernel<<<(int)((nE + 255) / 256), 256, 0, stream>>>(X, TGT, 1);
  // tgt_q = tgt + pos
  add_kernel<<<(int)((nE + 255) / 256), 256, 0, stream>>>(TGT, pos, TGTQ, nE);
  // deformable attention
  gemm(memory, val_W, val_b, nullptr, VALH, 272000, 256, 256, 0);
  gemm(TGTQ, off_W, off_b, OFFS, nullptr, M1, 256, 256, 0);
  gemm(TGTQ, aw_W, aw_b, AW, nullptr, M1, 128, 256, 0);
  softmax16_kernel<<<(108800 + 255) / 256, 256, 0, stream>>>(AW, 108800);
  deform_kernel<<<13600, 256, 0, stream>>>(refpts, OFFS, AW, VALH, ATT);
  gemm(ATT, outp_W, outp_b, T2, nullptr, M1, 256, 256, 0);
  // gate
  concat_kernel<<<(int)(((long)M1 * 512 + 255) / 256), 256, 0, stream>>>(TGTQ, T2, CAT, (long)M1 * 512);
  gemm(CAT, gate_W, gate_b, G, nullptr, M1, 512, 512, 2);
  ln_kernel<<<M1, 256, 0, stream>>>(TGTQ, T2, G, gate_ng, gate_nb, TGT, 1);
  // FFN
  gemm(TGT, ffn_W1, ffn_b1, H, nullptr, M1, 1024, 256, 1);
  gemm(H, ffn_W2, ffn_b2, X, nullptr, M1, 256, 1024, 0);
  ln_kernel<<<M1, 256, 0, stream>>>(TGT, X, nullptr, n2_g, n2_b, (float*)d_out, 2);
}
